// MyModel_87522843560469
// MI455X (gfx1250) — matrix operand order repaired and checked by reading, not compiled, not run on hardware
//
#include <hip/hip_runtime.h>
#include <hip/hip_bf16.h>
#include <math.h>

#define B_        2
#define N_        19248
#define NCLS_     81
#define C_        80
#define MD_       32
#define PH_       138
#define PW_       138
#define HW_       (PH_*PW_)        // 19044
#define MAXOUT_   300
#define MAXCLS_   100
#define IOU_THR_  0.5f
#define SCORE_THR_ 0.3f
// -0.5f / SIGMA  with SIGMA = 0.5  ->  -1.0
#define NMS_COEF_ (-1.0f)

#define M_TILES_  ((HW_ + 15) / 16)      // 1191
#define N_TILES_  ((MAXOUT_ + 15) / 16)  // 19

typedef __attribute__((ext_vector_type(16))) _Float16 v16h;
typedef __attribute__((ext_vector_type(8)))  float    v8f;

static __device__ __forceinline__ float sigm(float x) {
  return 1.0f / (1.0f + __expf(-x));
}

// ---------------------------------------------------------------- init flags
__global__ void k_init(int* __restrict__ any_pass) {
  if (threadIdx.x < B_) any_pass[threadIdx.x] = 0;
}

// --------------------------------------------------- decode boxes + softmax
// probs written class-major: probs[b][c][n] so each NMS block streams
// one contiguous, L2-resident row of N floats.
__global__ void k_decode_softmax(const float* __restrict__ off,
                                 const float* __restrict__ cls,
                                 const float* __restrict__ priors,
                                 float* __restrict__ boxes,
                                 float* __restrict__ probs,
                                 int* __restrict__ any_pass) {
  int tid = blockIdx.x * blockDim.x + threadIdx.x;
  if (tid >= B_ * N_) return;
  int b = tid / N_;
  int n = tid - b * N_;

  #pragma unroll
  for (int q = 0; q < 4; ++q)
    boxes[(size_t)tid * 4 + q] = off[(size_t)tid * 4 + q] + priors[(size_t)n * 4 + q];

  const float* l = cls + (size_t)tid * NCLS_;
  float m = -__builtin_inff();
  for (int c = 0; c < NCLS_; ++c) m = fmaxf(m, l[c]);
  float sum = 0.0f;
  for (int c = 0; c < NCLS_; ++c) sum += expf(l[c] - m);
  float inv = 1.0f / sum;

  float pmax = 0.0f;
  for (int c = 1; c < NCLS_; ++c) {
    float p = expf(l[c] - m) * inv;
    probs[((size_t)b * C_ + (c - 1)) * N_ + n] = p;
    pmax = fmaxf(pmax, p);
  }
  if (pmax > SCORE_THR_) atomicOr(&any_pass[b], 1);
}

// ----------------------------------------------------------------- soft-NMS
// One block per (class, batch): 100 sequential rounds of
// (parallel argmax over N) -> (Gaussian score decay). Scores are updated
// in place in the probs row; each thread only re-reads elements it wrote.
__global__ void k_softnms(const float* __restrict__ boxes,
                          float* __restrict__ probs,
                          float* __restrict__ nms_sc,
                          int* __restrict__ nms_idx,
                          int* __restrict__ nms_valid) {
  const int c = blockIdx.x;
  const int b = blockIdx.y;
  const int t = threadIdx.x;
  const int NT = blockDim.x;

  float* s = probs + ((size_t)b * C_ + c) * N_;
  const float* bx = boxes + (size_t)b * N_ * 4;

  __shared__ float rv[256];
  __shared__ int   ri[256];
  __shared__ int   sh_i;
  __shared__ int   sh_valid;

  const float NEG_INF = -__builtin_inff();

  for (int k = 0; k < MAXCLS_; ++k) {
    // --- parallel first-max argmax (tie -> lowest index) ---
    float bv = NEG_INF;
    int   bi = N_;
    for (int j = t; j < N_; j += NT) {
      float v = s[j];
      if (v > bv || (v == bv && j < bi)) { bv = v; bi = j; }
    }
    rv[t] = bv; ri[t] = bi;
    __syncthreads();
    for (int st = 128; st > 0; st >>= 1) {
      if (t < st) {
        float ov = rv[t + st]; int oi = ri[t + st];
        if (ov > rv[t] || (ov == rv[t] && oi < ri[t])) { rv[t] = ov; ri[t] = oi; }
      }
      __syncthreads();
    }
    if (t == 0) {
      size_t o = ((size_t)b * C_ + c) * MAXCLS_ + k;
      nms_idx[o]   = ri[0];
      nms_sc[o]    = rv[0];
      nms_valid[o] = (rv[0] > SCORE_THR_) ? 1 : 0;
      sh_i = ri[0]; sh_valid = (rv[0] > SCORE_THR_) ? 1 : 0;
    }
    __syncthreads();

    int i     = sh_i;
    int valid = sh_valid;

    if (valid) {
      float b0 = bx[(size_t)i * 4 + 0], b1 = bx[(size_t)i * 4 + 1];
      float b2 = bx[(size_t)i * 4 + 2], b3 = bx[(size_t)i * 4 + 3];
      float y1i = fminf(b0, b2), y2i = fmaxf(b0, b2);
      float x1i = fminf(b1, b3), x2i = fmaxf(b1, b3);
      float areai = (y2i - y1i) * (x2i - x1i);

      for (int j = t; j < N_; j += NT) {
        float c0 = bx[(size_t)j * 4 + 0], c1 = bx[(size_t)j * 4 + 1];
        float c2 = bx[(size_t)j * 4 + 2], c3 = bx[(size_t)j * 4 + 3];
        float y1 = fminf(c0, c2), y2 = fmaxf(c0, c2);
        float x1 = fminf(c1, c3), x2 = fmaxf(c1, c3);
        float area = (y2 - y1) * (x2 - x1);

        float iy1 = fmaxf(y1, y1i), ix1 = fmaxf(x1, x1i);
        float iy2 = fminf(y2, y2i), ix2 = fminf(x2, x2i);
        float inter = fmaxf(iy2 - iy1, 0.0f) * fmaxf(ix2 - ix1, 0.0f);
        float uni = area + areai - inter;
        float iou = ((area > 0.0f) && (areai > 0.0f))
                      ? inter / ((uni > 0.0f) ? uni : 1.0f) : 0.0f;
        float w = (iou <= IOU_THR_) ? expf(NMS_COEF_ * iou * iou) : 0.0f;
        float sn = (j == i) ? NEG_INF : s[j] * w;
        s[j] = sn;
      }
    }
    __syncthreads();
  }
}

// ----------------------------------- stable top-300 by exact rank counting
__global__ void k_topk(const float* __restrict__ boxes,
                       const float* __restrict__ coefs,
                       const float* __restrict__ nms_sc,
                       const int* __restrict__ nms_idx,
                       const int* __restrict__ nms_valid,
                       const int* __restrict__ any_pass,
                       float* __restrict__ ob,
                       float* __restrict__ ocls,
                       float* __restrict__ osc,
                       int* __restrict__ ond,
                       float* __restrict__ selcoef) {
  const int b = blockIdx.x;
  const int t = threadIdx.x;
  const int NT = blockDim.x;
  const int M = C_ * MAXCLS_;  // 8000 candidates

  __shared__ float fs[C_ * MAXCLS_];

  for (int j = t; j < M; j += NT) {
    size_t o = (size_t)b * M + j;
    fs[j] = nms_valid[o] ? nms_sc[o] : 0.0f;
  }
  __syncthreads();

  for (int j = t; j < M; j += NT) {
    float sj = fs[j];
    int r = 0;
    for (int k2 = 0; k2 < M; ++k2) {
      float sk = fs[k2];
      r += ((sk > sj) || (sk == sj && k2 < j)) ? 1 : 0;
    }
    if (r < MAXOUT_) {
      int c = j / MAXCLS_;
      size_t o = (size_t)b * M + j;
      int idx   = nms_idx[o];
      int valid = nms_valid[o];
      float vf  = valid ? 1.0f : 0.0f;
      #pragma unroll
      for (int q = 0; q < 4; ++q)
        ob[((size_t)b * MAXOUT_ + r) * 4 + q] =
            boxes[((size_t)b * N_ + idx) * 4 + q] * vf;
      ocls[(size_t)b * MAXOUT_ + r] = valid ? (float)(c + 1) : 0.0f;
      osc[(size_t)b * MAXOUT_ + r]  = sj;
      for (int d = 0; d < MD_; ++d)
        selcoef[((size_t)b * MAXOUT_ + r) * MD_ + d] =
            coefs[((size_t)b * N_ + idx) * MD_ + d] * vf;
    }
  }
  if (t == 0) ond[b] = any_pass[b] ? MAXOUT_ : 0;
}

// ------------------------------------- stage proto -> f16 A-tiles (WMMA layout)
// One thread per (b, mTile, lane): writes one contiguous 32B record holding
// the 16 halves that lane owns for V_WMMA_F32_16X16X32_F16's A operand.
// Loads are unconditional (clamped address) + value select -> no EXEC churn.
__global__ void k_stage_a(const float* __restrict__ proto,
                          _Float16* __restrict__ sa) {
  int tid = blockIdx.x * blockDim.x + threadIdx.x;  // B * M_TILES * 32
  if (tid >= B_ * M_TILES_ * 32) return;
  int lane = tid & 31;
  int mt   = (tid >> 5) % M_TILES_;
  int b    = tid / (M_TILES_ * 32);
  int mloc = lane & 15;
  int hi   = lane >> 4;
  int pix  = mt * 16 + mloc;
  int pc   = (pix < HW_) ? pix : (HW_ - 1);
  bool ok  = (pix < HW_);

  v16h a;
  #pragma unroll
  for (int h = 0; h < 16; ++h) {
    int k = ((h < 8) ? 0 : 16) + (hi ? 8 : 0) + (h & 7);
    float v = proto[((size_t)b * HW_ + pc) * MD_ + k];
    a[h] = (_Float16)(ok ? v : 0.0f);
  }
  *(v16h*)(sa + (size_t)tid * 16) = a;
}

// ---------------------------- stage selcoef -> f16 B-tiles (WMMA layout, padded)
__global__ void k_stage_b(const float* __restrict__ selcoef,
                          _Float16* __restrict__ sb) {
  int tid = blockIdx.x * blockDim.x + threadIdx.x;  // B * N_TILES * 32
  if (tid >= B_ * N_TILES_ * 32) return;
  int lane = tid & 31;
  int nt   = (tid >> 5) % N_TILES_;
  int b    = tid / (N_TILES_ * 32);
  int nloc = lane & 15;
  int hi   = lane >> 4;
  int col  = nt * 16 + nloc;
  int cc   = (col < MAXOUT_) ? col : (MAXOUT_ - 1);
  bool ok  = (col < MAXOUT_);

  v16h bm;
  #pragma unroll
  for (int h = 0; h < 16; ++h) {
    int k = ((h < 8) ? 0 : 16) + (hi ? 8 : 0) + (h & 7);
    float v = selcoef[((size_t)b * MAXOUT_ + cc) * MD_ + k];
    bm[h] = (_Float16)(ok ? v : 0.0f);
  }
  *(v16h*)(sb + (size_t)tid * 16) = bm;
}

// ------------------------------------------- mask GEMM + sigmoid via WMMA
// A (one 32B load, reused for all 19 N-tiles) and B (one 32B load per tile)
// are pre-staged in register layout; one v_wmma_f32_16x16x32_f16 per tile,
// fully unrolled. Stores are packed float4 pairs (rows per lane are 8
// contiguous, 16B-aligned floats); only the single last M-tile takes the
// guarded path (wave-uniform branch keeps EXEC all-1s at the WMMAs).
__global__ void k_masks(const _Float16* __restrict__ sa,
                        const _Float16* __restrict__ sb,
                        float* __restrict__ masks) {
  const int lane  = threadIdx.x & 31;
  const int wid   = threadIdx.x >> 5;
  const int mTile = blockIdx.x * 4 + wid;   // wave-uniform
  if (mTile >= M_TILES_) return;            // whole-wave exit
  const int b       = blockIdx.y;
  const int pixBase = mTile * 16;
  const int mloc    = lane & 15;
  const int hi      = lane >> 4;
  const int mOff    = hi ? 8 : 0;
  const bool fullTile = (pixBase + 16 <= HW_);  // wave-uniform

  const v16h a =
      *(const v16h*)(sa + ((size_t)(b * M_TILES_ + mTile) * 32 + lane) * 16);

  #pragma unroll
  for (int nt = 0; nt < N_TILES_; ++nt) {
    const v16h bm =
        *(const v16h*)(sb + ((size_t)(b * N_TILES_ + nt) * 32 + lane) * 16);

    v8f acc = {};
    acc = __builtin_amdgcn_wmma_f32_16x16x32_f16(
        /*neg_a=*/false, a, /*neg_b=*/false, bm,
        /*c_mod=*/(short)0, acc, /*reuse_a=*/false, /*reuse_b=*/false);

    const int col = nt * 16 + mloc;
    float* dst = masks + ((size_t)b * MAXOUT_ + col) * HW_ + pixBase + mOff;

    if (col < MAXOUT_) {
      if (fullTile) {
        float4 lo, hi4;
        lo.x  = sigm(acc[0]); lo.y  = sigm(acc[1]);
        lo.z  = sigm(acc[2]); lo.w  = sigm(acc[3]);
        hi4.x = sigm(acc[4]); hi4.y = sigm(acc[5]);
        hi4.z = sigm(acc[6]); hi4.w = sigm(acc[7]);
        *reinterpret_cast<float4*>(dst)     = lo;
        *reinterpret_cast<float4*>(dst + 4) = hi4;
      } else {
        #pragma unroll
        for (int r = 0; r < 8; ++r) {
          int p = pixBase + mOff + r;
          if (p < HW_) dst[r] = sigm(acc[r]);
        }
      }
    }
  }
}

// ---------------------------------------------------------------- launcher
extern "C" void kernel_launch(void* const* d_in, const int* in_sizes, int n_in,
                              void* d_out, int out_size, void* d_ws, size_t ws_size,
                              hipStream_t stream) {
  const float* pred_offset = (const float*)d_in[0];  // (B,N,4)
  const float* pred_cls    = (const float*)d_in[1];  // (B,N,81)
  const float* pred_coef   = (const float*)d_in[2];  // (B,N,32)
  const float* priors      = (const float*)d_in[3];  // (N,4)
  const float* proto       = (const float*)d_in[4];  // (B,138,138,32)

  // d_out: boxes (B,300,4) | classes (B,300) | scores (B,300) |
  //        masks (B,300,138,138) | num_det (B, int32 bits)
  float* out        = (float*)d_out;
  float* out_boxes  = out;
  float* out_cls    = out_boxes + (size_t)B_ * MAXOUT_ * 4;
  float* out_scores = out_cls + (size_t)B_ * MAXOUT_;
  float* out_masks  = out_scores + (size_t)B_ * MAXOUT_;
  int*   out_numdet = (int*)(out_masks + (size_t)B_ * MAXOUT_ * HW_);

  // workspace layout
  float* ws        = (float*)d_ws;
  float* boxes     = ws;                                  // B*N*4
  float* probs     = boxes + (size_t)B_ * N_ * 4;         // B*C*N
  float* nms_sc    = probs + (size_t)B_ * C_ * N_;        // B*C*100
  int*   nms_idx   = (int*)(nms_sc + (size_t)B_ * C_ * MAXCLS_);
  int*   nms_valid = nms_idx + (size_t)B_ * C_ * MAXCLS_;
  int*   any_pass  = nms_valid + (size_t)B_ * C_ * MAXCLS_;
  float* selcoef   = (float*)(any_pass + 8);              // B*300*32

  uintptr_t pa = (uintptr_t)(selcoef + (size_t)B_ * MAXOUT_ * MD_);
  pa = (pa + 31) & ~(uintptr_t)31;
  _Float16* stageA = (_Float16*)pa;                       // B*M_TILES*32*16
  _Float16* stageB = stageA + (size_t)B_ * M_TILES_ * 32 * 16;  // B*19*32*16

  k_init<<<1, 32, 0, stream>>>(any_pass);

  int nth = B_ * N_;
  k_decode_softmax<<<(nth + 255) / 256, 256, 0, stream>>>(
      pred_offset, pred_cls, priors, boxes, probs, any_pass);

  // A staging depends only on proto; overlap with NMS.
  int na = B_ * M_TILES_ * 32;
  k_stage_a<<<(na + 255) / 256, 256, 0, stream>>>(proto, stageA);

  dim3 g2(C_, B_);
  k_softnms<<<g2, 256, 0, stream>>>(boxes, probs, nms_sc, nms_idx, nms_valid);

  k_topk<<<B_, 1024, 0, stream>>>(boxes, pred_coef, nms_sc, nms_idx, nms_valid,
                                  any_pass, out_boxes, out_cls, out_scores,
                                  out_numdet, selcoef);

  int nb = B_ * N_TILES_ * 32;
  k_stage_b<<<(nb + 255) / 256, 256, 0, stream>>>(selcoef, stageB);

  dim3 g4((M_TILES_ + 3) / 4, B_);
  k_masks<<<g4, 128, 0, stream>>>(stageA, stageB, out_masks);

  (void)in_sizes; (void)n_in; (void)out_size; (void)ws_size;
}
